// PaTHAttention_76982993813768
// MI455X (gfx1250) — compile-verified, matrix operand order checked by reading, not run on hardware
//
#include <hip/hip_runtime.h>
#include <hip/hip_bf16.h>
#include <math.h>

// ---------------------------------------------------------------------------
// PaTH attention forward, MI455X (gfx1250, wave32, WMMA bf16)
// B=1, T=1024, HID=2048, NH=16, HD=128
// ---------------------------------------------------------------------------

#define T_LEN 1024
#define HID   2048
#define NH    16
#define HD    128
#define SCALE_ATTN 0.08838834764831845f  // 128^-0.5

typedef __bf16 bf16;
typedef __attribute__((ext_vector_type(16))) __bf16 v16bf;
typedef __attribute__((ext_vector_type(8)))  float  v8f;

__device__ __forceinline__ bf16 f2bf(float f) {
  union { float f; unsigned u; } x; x.f = f;
  unsigned r = x.u + 0x7fffu + ((x.u >> 16) & 1u);   // RNE
  union { unsigned short s; bf16 b; } y; y.s = (unsigned short)(r >> 16);
  return y.b;
}

// Load a 16x32 bf16 WMMA A/B fragment.
//   lanes 0-15  (row/col = lane)    hold K {0..7, 16..23}
//   lanes 16-31 (row/col = lane-16) hold K {8..15, 24..31}
// => two 16-byte contiguous loads per lane.
__device__ __forceinline__ v16bf load_frag(const bf16* __restrict__ base,
                                           int stride, int rc0, int k0) {
  int lane = threadIdx.x & 31;
  int half = lane >> 4;
  int rc   = lane & 15;
  const bf16* p = base + (size_t)(rc0 + rc) * stride + k0 + half * 8;
  union { uint4 u[2]; v16bf v; } t;
  t.u[0] = *(const uint4*)(p);
  t.u[1] = *(const uint4*)(p + 16);
  return t.v;
}

__device__ __forceinline__ v8f wmma_bf16(v16bf a, v16bf b, v8f c) {
  return __builtin_amdgcn_wmma_f32_16x16x32_bf16(false, a, false, b,
                                                 (short)0, c, false, false);
}

// C/D tile layout: value acc[r] of lane L sits at (M = r + (L>>4)*8, N = L&15).

// ---------------------------------------------------------------------------
// elementwise / conversion kernels
// ---------------------------------------------------------------------------

__global__ __launch_bounds__(256) void k_cvt_bf16(const float* __restrict__ in,
                                                  bf16* __restrict__ out, int n) {
  int i = blockIdx.x * blockDim.x + threadIdx.x;
  if (i < n) out[i] = f2bf(in[i]);
}

// in [R][C] f32 -> out [C][R] bf16
__global__ __launch_bounds__(256) void k_transpose_cvt(const float* __restrict__ in,
                                                       bf16* __restrict__ out,
                                                       int R, int C) {
  int i = blockIdx.x * blockDim.x + threadIdx.x;
  if (i >= R * C) return;
  int r = i / C, c = i % C;
  out[(size_t)c * R + r] = f2bf(in[i]);
}

__global__ __launch_bounds__(256) void k_zero_bf16(bf16* __restrict__ p, size_t n) {
  size_t i = (size_t)blockIdx.x * blockDim.x + threadIdx.x;
  size_t stride = (size_t)gridDim.x * blockDim.x;
  union { unsigned short s; bf16 b; } z; z.s = 0;
  for (; i < n; i += stride) p[i] = z.b;
}

// RMSNorm row t of [T][HID] f32, write head-major bf16 [NH][T][HD]
__global__ __launch_bounds__(256) void k_rmsnorm_heads(const float* __restrict__ in,
                                                       const float* __restrict__ wgt,
                                                       bf16* __restrict__ out) {
  int t = blockIdx.x;
  __shared__ float red[256];
  float s = 0.f;
  for (int c = threadIdx.x; c < HID; c += 256) {
    float v = in[(size_t)t * HID + c]; s += v * v;
  }
  red[threadIdx.x] = s; __syncthreads();
  for (int st = 128; st > 0; st >>= 1) {
    if (threadIdx.x < st) red[threadIdx.x] += red[threadIdx.x + st];
    __syncthreads();
  }
  float sc = rsqrtf(red[0] / (float)HID + 1e-6f);
  for (int c = threadIdx.x; c < HID; c += 256) {
    int h = c >> 7, d = c & 127;
    out[((size_t)h * T_LEN + t) * HD + d] =
        f2bf(in[(size_t)t * HID + c] * sc * wgt[c]);
  }
}

// causal depthwise conv (K=3) + per-head l2norm; out head-major bf16
__global__ __launch_bounds__(128) void k_conv_l2_heads(const float* __restrict__ wpre,
                                                       const float* __restrict__ convw,
                                                       bf16* __restrict__ out) {
  int t = blockIdx.x, h = blockIdx.y, d = threadIdx.x;  // blockDim 128
  int c = h * HD + d;
  float acc = 0.f;
  #pragma unroll
  for (int i = 0; i < 3; ++i) {
    int tt = t + i - 2;
    if (tt >= 0) acc += wpre[(size_t)tt * HID + c] * convw[c * 3 + i];
  }
  __shared__ float red[128];
  red[d] = acc * acc; __syncthreads();
  for (int st = 64; st > 0; st >>= 1) {
    if (d < st) red[d] += red[d + st];
    __syncthreads();
  }
  float sc = rsqrtf(red[0] + 1e-6f);
  out[((size_t)h * T_LEN + t) * HD + d] = f2bf(acc * sc);
}

// v [T][HID] f32 -> vt [NH][HD][T] bf16 (time contiguous: B-fragment for P@V)
__global__ __launch_bounds__(256) void k_v_transpose(const float* __restrict__ v,
                                                     bf16* __restrict__ vt) {
  int i = blockIdx.x * blockDim.x + threadIdx.x;
  if (i >= NH * HD * T_LEN) return;
  int t = i % T_LEN; int rest = i / T_LEN;
  int d = rest & 127; int h = rest >> 7;
  vt[i] = f2bf(v[(size_t)t * HID + h * HD + d]);
}

// gates: beta = 2*sigmoid(btraw+bbt), g = logsigmoid(graw+bg); head-major [NH][T]
// btraw/graw are [T][128] padded (stride 128)
__global__ __launch_bounds__(256) void k_gate(const float* __restrict__ btraw,
                                              const float* __restrict__ graw,
                                              const float* __restrict__ bbt,
                                              const float* __restrict__ bg,
                                              float* __restrict__ beta,
                                              float* __restrict__ g) {
  int i = blockIdx.x * blockDim.x + threadIdx.x;
  if (i >= NH * T_LEN) return;
  int t = i % T_LEN, h = i / T_LEN;
  float xb = btraw[t * 128 + h] + bbt[h];
  beta[i] = 2.f / (1.f + __expf(-xb));
  float xg = graw[t * 128 + h] + bg[h];
  g[i] = fminf(xg, 0.f) - log1pf(__expf(-fabsf(xg)));
}

__global__ __launch_bounds__(32) void k_cumsum(const float* __restrict__ g,
                                               float* __restrict__ cg) {
  int h = threadIdx.x;
  if (h >= NH) return;
  float s = 0.f;
  for (int t = 0; t < T_LEN; ++t) { s += g[h * T_LEN + t]; cg[h * T_LEN + t] = s; }
}

// ---------------------------------------------------------------------------
// generic GEMM:  C[M][N] f32 (ldc) = A[M][K] bf16 (lda) @ Bt[N][K]^T bf16 (ldb)
// block = 256 threads = 8 waves (2x4), tile 64(M) x 128(N).
// Each wave: 2x2 register tile (32x32), K chunks of 32, double-buffered
// fragment loads so global loads overlap the WMMAs of the previous chunk.
// No guards: callers pad all operands so every tile is valid.
// ---------------------------------------------------------------------------
__global__ __launch_bounds__(256) void k_gemm_nt(const bf16* __restrict__ A, int lda,
                                                 const bf16* __restrict__ Bt, int ldb,
                                                 float* __restrict__ C, int ldc,
                                                 int K) {
  int wave = threadIdx.x >> 5;
  int wm = wave & 1, wn = wave >> 1;                // 2 x 4 waves
  int m0 = blockIdx.y * 64 + wm * 32;
  int n0 = blockIdx.x * 128 + wn * 32;
  v8f c00 = {}, c01 = {}, c10 = {}, c11 = {};

  v16bf a0 = load_frag(A, lda, m0, 0);
  v16bf a1 = load_frag(A, lda, m0 + 16, 0);
  v16bf b0 = load_frag(Bt, ldb, n0, 0);
  v16bf b1 = load_frag(Bt, ldb, n0 + 16, 0);

  for (int k0 = 0; k0 < K; k0 += 32) {
    v16bf na0 = a0, na1 = a1, nb0 = b0, nb1 = b1;
    int kn = k0 + 32;
    if (kn < K) {                                   // scalar (uniform) branch
      na0 = load_frag(A, lda, m0, kn);
      na1 = load_frag(A, lda, m0 + 16, kn);
      nb0 = load_frag(Bt, ldb, n0, kn);
      nb1 = load_frag(Bt, ldb, n0 + 16, kn);
      __builtin_prefetch(A + (size_t)m0 * lda + kn + 32, 0, 1);
    }
    c00 = wmma_bf16(a0, b0, c00);
    c01 = wmma_bf16(a0, b1, c01);
    c10 = wmma_bf16(a1, b0, c10);
    c11 = wmma_bf16(a1, b1, c11);
    a0 = na0; a1 = na1; b0 = nb0; b1 = nb1;
  }

  int lane = threadIdx.x & 31;
  int nc = lane & 15, rh = (lane >> 4) * 8;
  #pragma unroll
  for (int r = 0; r < 8; ++r) {
    C[(size_t)(m0 + rh + r) * ldc + n0 + nc]           = c00[r];
    C[(size_t)(m0 + rh + r) * ldc + n0 + 16 + nc]      = c01[r];
    C[(size_t)(m0 + 16 + rh + r) * ldc + n0 + nc]      = c10[r];
    C[(size_t)(m0 + 16 + rh + r) * ldc + n0 + 16 + nc] = c11[r];
  }
}

// ---------------------------------------------------------------------------
// UT-transform solve, one workgroup per head:
//   (I + strict_tril(W W^T diag(beta))) A = strict_tril(W K^T)
// A stored TRANSPOSED bf16: At[c][m] (m contiguous) => usable as B-fragment.
// ---------------------------------------------------------------------------
__global__ __launch_bounds__(256) void k_solve(const bf16* __restrict__ wh,
                                               const bf16* __restrict__ kh,
                                               const float* __restrict__ beta_all,
                                               bf16* __restrict__ At_all) {
  int h = blockIdx.x;
  const bf16* w = wh + (size_t)h * T_LEN * HD;
  const bf16* k = kh + (size_t)h * T_LEN * HD;
  bf16* A = At_all + (size_t)h * T_LEN * T_LEN;
  const float* bet = beta_all + h * T_LEN;

  extern __shared__ char smem[];
  float* rhs   = (float*)smem;                 // [16][T_LEN]
  bf16*  lch   = (bf16*)(rhs + 16 * T_LEN);    // [16][32]
  float* ldiag = (float*)(lch + 16 * 32);      // [16][16]

  int tid = threadIdx.x, wave = tid >> 5, lane = tid & 31;
  int nc = lane & 15, rh = (lane >> 4) * 8;

  for (int ib = 0; ib < T_LEN / 16; ++ib) {
    int r0 = ib * 16;
    int ncols = r0 + 16;

    // hoisted W row-block fragments (used by phases A, B, C)
    v16bf wf0 = load_frag(w, HD, r0, 0);
    v16bf wf1 = load_frag(w, HD, r0, 32);
    v16bf wf2 = load_frag(w, HD, r0, 64);
    v16bf wf3 = load_frag(w, HD, r0, 96);

    // phase A: rhs = strict_tril(w_rows . k^T)
    for (int jt = wave; jt <= ib; jt += 8) {
      int c0 = jt * 16;
      v8f acc = {};
      acc = wmma_bf16(wf0, load_frag(k, HD, c0, 0),  acc);
      acc = wmma_bf16(wf1, load_frag(k, HD, c0, 32), acc);
      acc = wmma_bf16(wf2, load_frag(k, HD, c0, 64), acc);
      acc = wmma_bf16(wf3, load_frag(k, HD, c0, 96), acc);
      #pragma unroll
      for (int r = 0; r < 8; ++r) {
        int row = rh + r;
        float vv = acc[r];
        if (c0 + nc >= r0 + row) vv = 0.f;      // strictly lower
        rhs[row * T_LEN + c0 + nc] = vv;
      }
    }
    __syncthreads();

    // phase B: rhs -= L[rows, m<r0] @ A[m, cols]
    for (int mc = 0; mc < r0; mc += 32) {
      if (wave < 2) {                            // build L chunk [16][32] bf16
        int m0 = mc + wave * 16;
        v8f acc = {};
        acc = wmma_bf16(wf0, load_frag(w, HD, m0, 0),  acc);
        acc = wmma_bf16(wf1, load_frag(w, HD, m0, 32), acc);
        acc = wmma_bf16(wf2, load_frag(w, HD, m0, 64), acc);
        acc = wmma_bf16(wf3, load_frag(w, HD, m0, 96), acc);
        #pragma unroll
        for (int r = 0; r < 8; ++r) {
          int m = m0 + nc;
          float vv = (m < r0) ? acc[r] * bet[m] : 0.f;
          lch[(rh + r) * 32 + wave * 16 + nc] = f2bf(vv);
        }
      }
      __syncthreads();
      int jhi = min(ib, (mc + 31) >> 4);
      for (int jt = wave; jt <= jhi; jt += 8) {
        int c0 = jt * 16;
        v8f acc = {};
        v16bf a = load_frag(lch, 32, 0, 0);
        v16bf b = load_frag(A, T_LEN, c0, mc);
        acc = wmma_bf16(a, b, acc);
        #pragma unroll
        for (int r = 0; r < 8; ++r)
          rhs[(rh + r) * T_LEN + c0 + nc] -= acc[r];
      }
      __syncthreads();
    }

    // phase C: diagonal 16x16 unit-lower solve
    if (wave == 0) {
      v8f acc = {};
      acc = wmma_bf16(wf0, load_frag(w, HD, r0, 0),  acc);
      acc = wmma_bf16(wf1, load_frag(w, HD, r0, 32), acc);
      acc = wmma_bf16(wf2, load_frag(w, HD, r0, 64), acc);
      acc = wmma_bf16(wf3, load_frag(w, HD, r0, 96), acc);
      #pragma unroll
      for (int r = 0; r < 8; ++r) {
        int row = rh + r, s = nc;
        ldiag[row * 16 + s] = (s < row) ? acc[r] * bet[r0 + s] : 0.f;
      }
    }
    __syncthreads();
    for (int r = 0; r < 16; ++r) {
      for (int s = r + 1; s < 16; ++s) {
        float lv = ldiag[s * 16 + r];
        for (int c = tid; c < ncols; c += 256)
          rhs[s * T_LEN + c] -= lv * rhs[r * T_LEN + c];
      }
      __syncthreads();
    }

    // write A rows (transposed: At[c][R])
    for (int r = 0; r < 16; ++r) {
      int R = r0 + r;
      for (int c = tid; c < R; c += 256)
        A[(size_t)c * T_LEN + R] = f2bf(rhs[r * T_LEN + c]);
    }
    __threadfence();
    __syncthreads();
  }
}

// ---------------------------------------------------------------------------
// attention: per (head, 16-row block)
//   S = (QK^T - tril(QW^T diag(beta)) @ A) * scale + cg_i - cg_j, causal softmax,
//   O = P V
// ---------------------------------------------------------------------------
__global__ __launch_bounds__(256) void k_attn(const bf16* __restrict__ qh,
                                              const bf16* __restrict__ kh,
                                              const bf16* __restrict__ wh,
                                              const bf16* __restrict__ vt,
                                              const bf16* __restrict__ At_all,
                                              const float* __restrict__ beta_all,
                                              const float* __restrict__ cg_all,
                                              float* __restrict__ o) {
  int h = blockIdx.y, ib = blockIdx.x;
  int r0 = ib * 16;
  const bf16* q = qh + (size_t)h * T_LEN * HD;
  const bf16* k = kh + (size_t)h * T_LEN * HD;
  const bf16* w = wh + (size_t)h * T_LEN * HD;
  const bf16* v = vt + (size_t)h * HD * T_LEN;
  const bf16* A = At_all + (size_t)h * T_LEN * T_LEN;
  const float* bet = beta_all + h * T_LEN;
  const float* cgh = cg_all + h * T_LEN;

  extern __shared__ char smem[];
  float* S    = (float*)smem;                      // [16][T_LEN] f32
  bf16*  P    = (bf16*)(S + 16 * T_LEN);           // [16][T_LEN] bf16
  bf16*  dch  = (bf16*)(P + 16 * T_LEN);           // [16][32]
  float* red  = (float*)(dch + 16 * 32);           // [16][16]
  float* rowm = red + 16 * 16;                     // [16]
  float* rows = rowm + 16;                         // [16]

  int tid = threadIdx.x, wave = tid >> 5, lane = tid & 31;
  int nc = lane & 15, rh = (lane >> 4) * 8;
  int ncols = r0 + 16;

  // hoisted Q row-block fragments (used by phases 1 and 2)
  v16bf qf0 = load_frag(q, HD, r0, 0);
  v16bf qf1 = load_frag(q, HD, r0, 32);
  v16bf qf2 = load_frag(q, HD, r0, 64);
  v16bf qf3 = load_frag(q, HD, r0, 96);

  // phase 1: S = Q K^T
  for (int jt = wave; jt <= ib; jt += 8) {
    int c0 = jt * 16;
    v8f acc = {};
    acc = wmma_bf16(qf0, load_frag(k, HD, c0, 0),  acc);
    acc = wmma_bf16(qf1, load_frag(k, HD, c0, 32), acc);
    acc = wmma_bf16(qf2, load_frag(k, HD, c0, 64), acc);
    acc = wmma_bf16(qf3, load_frag(k, HD, c0, 96), acc);
    #pragma unroll
    for (int r = 0; r < 8; ++r) S[(rh + r) * T_LEN + c0 + nc] = acc[r];
  }
  __syncthreads();

  // phase 2: S -= D @ A, D built on the fly in 32-wide m chunks
  for (int mc = 0; mc < ncols; mc += 32) {
    if (wave < 2) {
      int m0 = mc + wave * 16;
      v8f acc = {};
      acc = wmma_bf16(qf0, load_frag(w, HD, m0, 0),  acc);
      acc = wmma_bf16(qf1, load_frag(w, HD, m0, 32), acc);
      acc = wmma_bf16(qf2, load_frag(w, HD, m0, 64), acc);
      acc = wmma_bf16(qf3, load_frag(w, HD, m0, 96), acc);
      #pragma unroll
      for (int r = 0; r < 8; ++r) {
        int m = m0 + nc, row = r0 + rh + r;
        float vv = (m <= row) ? acc[r] * bet[m] : 0.f;   // tril incl diagonal
        dch[(rh + r) * 32 + wave * 16 + nc] = f2bf(vv);
      }
    }
    __syncthreads();
    int jhi = min(ib, (mc + 31) >> 4);
    for (int jt = wave; jt <= jhi; jt += 8) {
      int c0 = jt * 16;
      v8f acc = {};
      v16bf a = load_frag(dch, 32, 0, 0);
      v16bf b = load_frag(A, T_LEN, c0, mc);
      acc = wmma_bf16(a, b, acc);
      #pragma unroll
      for (int r = 0; r < 8; ++r)
        S[(rh + r) * T_LEN + c0 + nc] -= acc[r];
    }
    __syncthreads();
  }

  // phase 3: scale + gate + causal mask
  for (int r = 0; r < 16; ++r) {
    int R = r0 + r; float cgR = cgh[R];
    for (int c = tid; c < ncols; c += 256) {
      float s = (c <= R) ? (S[r * T_LEN + c] * SCALE_ATTN + cgR - cgh[c]) : -INFINITY;
      S[r * T_LEN + c] = s;
    }
  }
  __syncthreads();

  // softmax (16 threads / row)
  int row = tid >> 4, cix = tid & 15;
  float mx = -INFINITY;
  for (int c = cix; c < ncols; c += 16) mx = fmaxf(mx, S[row * T_LEN + c]);
  red[row * 16 + cix] = mx; __syncthreads();
  if (cix == 0) {
    float m2 = -INFINITY;
    for (int i2 = 0; i2 < 16; ++i2) m2 = fmaxf(m2, red[row * 16 + i2]);
    rowm[row] = m2;
  }
  __syncthreads();
  float rm = rowm[row];
  float sm = 0.f;
  for (int c = cix; c < ncols; c += 16) sm += __expf(S[row * T_LEN + c] - rm);
  red[row * 16 + cix] = sm; __syncthreads();
  if (cix == 0) {
    float s2 = 0.f;
    for (int i2 = 0; i2 < 16; ++i2) s2 += red[row * 16 + i2];
    rows[row] = s2;
  }
  __syncthreads();
  float inv = 1.f / rows[row];
  int padded = (ncols + 31) & ~31;
  for (int c = cix; c < padded; c += 16) {
    float pv = (c < ncols) ? __expf(S[row * T_LEN + c] - rm) * inv : 0.f;
    P[row * T_LEN + c] = f2bf(pv);
  }
  __syncthreads();

  // phase 4: O = P V  (8 waves, one 16-wide d-tile each)
  {
    int d0 = wave * 16;
    v8f acc = {};
    for (int kc = 0; kc < padded; kc += 32) {
      v16bf a = load_frag(P, T_LEN, 0, kc);
      v16bf b = load_frag(v, T_LEN, d0, kc);
      acc = wmma_bf16(a, b, acc);
    }
    #pragma unroll
    for (int r = 0; r < 8; ++r)
      o[(size_t)(r0 + rh + r) * HID + h * HD + d0 + nc] = acc[r];
  }
}

// ---------------------------------------------------------------------------
// host launcher
// ---------------------------------------------------------------------------
extern "C" void kernel_launch(void* const* d_in, const int* in_sizes, int n_in,
                              void* d_out, int out_size, void* d_ws, size_t ws_size,
                              hipStream_t stream) {
  const float* x     = (const float*)d_in[0];
  const float* Wq    = (const float*)d_in[1];
  const float* Wk    = (const float*)d_in[2];
  const float* Wv    = (const float*)d_in[3];
  const float* Ww1   = (const float*)d_in[4];
  const float* Ww2   = (const float*)d_in[5];
  const float* Wbt   = (const float*)d_in[6];
  const float* bbt   = (const float*)d_in[7];
  const float* Wg    = (const float*)d_in[8];
  const float* bg    = (const float*)d_in[9];
  const float* qnw   = (const float*)d_in[10];
  const float* knw   = (const float*)d_in[11];
  const float* convw = (const float*)d_in[12];
  const float* Wo    = (const float*)d_in[13];
  float* out = (float*)d_out;

  char* ws = (char*)d_ws;
  size_t off = 0;
  auto alloc = [&](size_t bytes) -> void* {
    void* p = ws + off;
    off += (bytes + 255) & ~(size_t)255;
    return p;
  };

  // padded to 128 rows / 128 cols where the logical N is 16 or 32, so the GEMM
  // kernel never needs guards (padding columns are computed and ignored).
  bf16* WqT  = (bf16*)alloc((size_t)HID * HID * 2);
  bf16* WkT  = (bf16*)alloc((size_t)HID * HID * 2);
  bf16* WvT  = (bf16*)alloc((size_t)HID * HID * 2);
  bf16* WoT  = (bf16*)alloc((size_t)HID * HID * 2);
  bf16* Ww1T = (bf16*)alloc((size_t)128 * HID * 2);   // logical 32 rows
  bf16* Ww2T = (bf16*)alloc((size_t)HID * 32 * 2);
  bf16* WbtT = (bf16*)alloc((size_t)128 * HID * 2);   // logical 16 rows
  bf16* WgT  = (bf16*)alloc((size_t)128 * HID * 2);   // logical 16 rows
  bf16* xbf  = (bf16*)alloc((size_t)T_LEN * HID * 2);
  float* qf  = (float*)alloc((size_t)T_LEN * HID * 4);
  float* kf  = (float*)alloc((size_t)T_LEN * HID * 4);
  float* vf  = (float*)alloc((size_t)T_LEN * HID * 4);
  float* xw1 = (float*)alloc((size_t)T_LEN * 128 * 4); // logical 32 cols
  bf16* xw1b = (bf16*)alloc((size_t)T_LEN * 128 * 2);
  float* wpre= (float*)alloc((size_t)T_LEN * HID * 4);
  float* btraw=(float*)alloc((size_t)T_LEN * 128 * 4); // logical 16 cols
  float* graw= (float*)alloc((size_t)T_LEN * 128 * 4); // logical 16 cols
  float* beta= (float*)alloc((size_t)NH * T_LEN * 4);
  float* gbuf= (float*)alloc((size_t)NH * T_LEN * 4);
  float* cg  = (float*)alloc((size_t)NH * T_LEN * 4);
  bf16* qhb  = (bf16*)alloc((size_t)NH * T_LEN * HD * 2);
  bf16* khb  = (bf16*)alloc((size_t)NH * T_LEN * HD * 2);
  bf16* whb  = (bf16*)alloc((size_t)NH * T_LEN * HD * 2);
  bf16* vtb  = (bf16*)alloc((size_t)NH * HD * T_LEN * 2);
  bf16* Atb  = (bf16*)alloc((size_t)NH * T_LEN * T_LEN * 2);
  float* of  = (float*)alloc((size_t)T_LEN * HID * 4);
  bf16* ofb  = (bf16*)alloc((size_t)T_LEN * HID * 2);

  dim3 blk(256);

  // convert / transpose
  k_cvt_bf16<<<(T_LEN * HID + 255) / 256, blk, 0, stream>>>(x, xbf, T_LEN * HID);
  k_transpose_cvt<<<(HID * HID + 255) / 256, blk, 0, stream>>>(Wq, WqT, HID, HID);
  k_transpose_cvt<<<(HID * HID + 255) / 256, blk, 0, stream>>>(Wk, WkT, HID, HID);
  k_transpose_cvt<<<(HID * HID + 255) / 256, blk, 0, stream>>>(Wv, WvT, HID, HID);
  k_transpose_cvt<<<(HID * HID + 255) / 256, blk, 0, stream>>>(Wo, WoT, HID, HID);
  k_transpose_cvt<<<(HID * 32 + 255) / 256, blk, 0, stream>>>(Ww1, Ww1T, HID, 32);
  k_transpose_cvt<<<(32 * HID + 255) / 256, blk, 0, stream>>>(Ww2, Ww2T, 32, HID);
  k_transpose_cvt<<<(HID * NH + 255) / 256, blk, 0, stream>>>(Wbt, WbtT, HID, NH);
  k_transpose_cvt<<<(HID * NH + 255) / 256, blk, 0, stream>>>(Wg, WgT, HID, NH);

  // projections: grid = (N/128, M/64)
  dim3 gBig(HID / 128, T_LEN / 64);
  dim3 gNarrow(1, T_LEN / 64);
  k_gemm_nt<<<gBig, blk, 0, stream>>>(xbf, HID, WqT, HID, qf, HID, HID);
  k_gemm_nt<<<gBig, blk, 0, stream>>>(xbf, HID, WkT, HID, kf, HID, HID);
  k_gemm_nt<<<gBig, blk, 0, stream>>>(xbf, HID, WvT, HID, vf, HID, HID);
  k_gemm_nt<<<gNarrow, blk, 0, stream>>>(xbf, HID, Ww1T, HID, xw1, 128, HID);
  k_gemm_nt<<<gNarrow, blk, 0, stream>>>(xbf, HID, WbtT, HID, btraw, 128, HID);
  k_gemm_nt<<<gNarrow, blk, 0, stream>>>(xbf, HID, WgT, HID, graw, 128, HID);

  // w = (x@Ww1)@Ww2  (xw1b has stride 128, logical K=32)
  k_cvt_bf16<<<(T_LEN * 128 + 255) / 256, blk, 0, stream>>>(xw1, xw1b, T_LEN * 128);
  k_gemm_nt<<<gBig, blk, 0, stream>>>(xw1b, 128, Ww2T, 32, wpre, HID, 32);

  // gates
  k_gate<<<(NH * T_LEN + 255) / 256, blk, 0, stream>>>(btraw, graw, bbt, bg, beta, gbuf);
  k_cumsum<<<1, 32, 0, stream>>>(gbuf, cg);

  // norms / conv / transposes into head-major bf16
  k_rmsnorm_heads<<<T_LEN, blk, 0, stream>>>(qf, qnw, qhb);
  k_rmsnorm_heads<<<T_LEN, blk, 0, stream>>>(kf, knw, khb);
  k_conv_l2_heads<<<dim3(T_LEN, NH), dim3(128), 0, stream>>>(wpre, convw, whb);
  k_v_transpose<<<(NH * HD * T_LEN + 255) / 256, blk, 0, stream>>>(vf, vtb);

  // UT solve -> A^T (zero-init so strict-upper reads are zero)
  k_zero_bf16<<<4096, blk, 0, stream>>>(Atb, (size_t)NH * T_LEN * T_LEN);
  size_t lds_solve = (size_t)16 * T_LEN * 4 + 16 * 32 * 2 + 16 * 16 * 4;
  k_solve<<<NH, blk, lds_solve, stream>>>(whb, khb, beta, Atb);

  // attention
  size_t lds_attn = (size_t)16 * T_LEN * 4 + (size_t)16 * T_LEN * 2 +
                    16 * 32 * 2 + 16 * 16 * 4 + 2 * 16 * 4;
  k_attn<<<dim3(T_LEN / 16, NH), blk, lds_attn, stream>>>(
      qhb, khb, whb, vtb, Atb, beta, cg, of);

  // output projection
  k_cvt_bf16<<<(T_LEN * HID + 255) / 256, blk, 0, stream>>>(of, ofb, T_LEN * HID);
  k_gemm_nt<<<gBig, blk, 0, stream>>>(ofb, HID, WoT, HID, out, HID, HID);
}
